// TransformerModel_75007308858108
// MI455X (gfx1250) — matrix operand order repaired and checked by reading, not compiled, not run on hardware
//
#include <hip/hip_runtime.h>

// ---------------------------------------------------------------------------
// Model constants (fixed by the reference setup_inputs)
// ---------------------------------------------------------------------------
constexpr int BB  = 2;      // batch
constexpr int SEQ = 512;    // sequence length
constexpr int DM  = 1024;   // model dim
constexpr int DFF = 4096;   // ffn dim
constexpr int VOC = 4096;   // vocab
constexpr int NH  = 16;     // heads
constexpr int DH  = 64;     // head dim
constexpr int NL  = 6;      // layers
constexpr int MTOK = BB * SEQ;   // 1024 token rows

// ---------------------------------------------------------------------------
// Vector types (wave32, gfx1250)
// ---------------------------------------------------------------------------
typedef __bf16    v16bf __attribute__((ext_vector_type(16)));
typedef __bf16    v8bf  __attribute__((ext_vector_type(8)));
typedef float     v8f   __attribute__((ext_vector_type(8)));
typedef float     v4f   __attribute__((ext_vector_type(4)));
typedef unsigned  u32x4 __attribute__((ext_vector_type(4)));
typedef int       i32x4 __attribute__((ext_vector_type(4)));
typedef int       i32x8 __attribute__((ext_vector_type(8)));

union AFrag { v16bf v; v8bf h[2]; };

#if __has_builtin(__builtin_amdgcn_tensor_load_to_lds)
#define HAVE_TDM 1
#endif

__device__ __forceinline__ __bf16 f2bf(float f) {
    unsigned u = __builtin_bit_cast(unsigned, f);
    unsigned r = u + 0x7FFFu + ((u >> 16) & 1u);   // round-to-nearest-even
    unsigned short h = (unsigned short)(r >> 16);
    return __builtin_bit_cast(__bf16, h);
}

// Epilogue modes
enum {
    EPI_BIAS = 0,       // fp32 out + bf16 dual write (+bias)
    EPI_BIAS_RELU,      // bf16 out = relu(acc + bias[n])
    EPI_BIAS_RES,       // fp32 out = acc + bias[n] + res[m,n]
    EPI_QKHEAD,         // fp32 scatter to [b,h,s,dh] (+bias)
    EPI_VTHEAD,         // fp32 scatter to [b,h,dh,s] (+bias)  (V transposed)
    EPI_SCALE,          // fp32 out = acc * alpha               (attn scores)
    EPI_PLAIN,          // bf16 out = acc                       (attn output)
    EPI_MASK            // fp32 out = (acc!=0) ? res[m,n] : -1e10
};

// ---------------------------------------------------------------------------
// Batched NT GEMM:  C[z] = A[z] (MxK, row-major) * W[z]^T (W is NxK, row-major)
// B side: fp32 weights streamed from HBM, converted to bf16 into LDS (each
//         weight byte crosses HBM exactly once -> bandwidth optimal).
// A side: either fp32 (convert path) or pre-converted bf16 activations DMA'd
//         into LDS by the Tensor Data Mover (ABF=true).
// 256 threads = 8 waves arranged 2(M) x 4(N). BK = 32 = one WMMA K step.
// Requires M%BM==0, N%BN==0, K%BK==0 (all true for this model).
// ---------------------------------------------------------------------------
template <int BM, int BN, int EPI, bool ABF>
__launch_bounds__(256)
__global__ void gemm_nt_wmma(const float* __restrict__ A,
                             const __bf16* __restrict__ Abf, int lda, long long sA,
                             const float* __restrict__ W, int ldw, long long sW,
                             const float* __restrict__ bias, int biasOff,
                             float* __restrict__ Out, __bf16* __restrict__ OutB,
                             int ldo,
                             int batchInner, long long soOuter, long long soInner,
                             const float* __restrict__ Res, int ldr,
                             int M, int N, int K, float alpha)
{
    constexpr int BK  = 32;
    constexpr int WM  = BM / 2;
    constexpr int WN  = BN / 4;
    constexpr int MI  = WM / 16;
    constexpr int NI  = WN / 16;
    constexpr int KV4 = BK / 4;                      // float4 chunks per row
    constexpr int ANV = BM * BK / (256 * 4);         // float4 loads/thread (A f32)
    constexpr int BNV = BN * BK / (256 * 4);         // float4 loads/thread (B)
    constexpr int ACH = BM * BK / (256 * 8);         // 16B bf16 chunks/thread (A bf16)
    static_assert(MI >= 1 && NI >= 1, "tile too small");

    __shared__ __align__(64) __bf16 As[BM][BK];
    __shared__ __align__(64) __bf16 Bs[BN][BK];

    const int z = blockIdx.z;
    const float*  Ab   = ABF ? nullptr : (A + (long long)z * sA);
    const __bf16* Abfb = ABF ? (Abf + (long long)z * sA) : nullptr;
    const float*  Wb   = W + (long long)z * sW;
    const long long obase = (long long)(z / batchInner) * soOuter
                          + (long long)(z % batchInner) * soInner;

    const int tid  = threadIdx.x;
    const int wave = tid >> 5;
    const int lane = tid & 31;
    const int wm   = wave >> 2;          // 0..1
    const int wn   = wave & 3;           // 0..3
    const int m0   = blockIdx.x * BM;
    const int n0   = blockIdx.y * BN;
    const int lm   = lane & 15;
    const int lg   = lane >> 4;
    const int kbA  = lg * 16;            // A fragment K base
    const int kbB  = lg * 16;            // B fragment K base

    v8f acc[MI][NI];
    for (int mi = 0; mi < MI; ++mi)
        for (int ni = 0; ni < NI; ++ni)
            for (int e = 0; e < 8; ++e) acc[mi][ni][e] = 0.0f;

    for (int k0 = 0; k0 < K; k0 += BK) {
        // ---- A tile -> LDS ----
        if constexpr (ABF) {
#ifdef HAVE_TDM
            // Tensor Data Mover: DMA the BMxBK bf16 tile straight into LDS.
            if (tid == 0) {
                unsigned long long ga =
                    (unsigned long long)(size_t)(Abfb + (long long)m0 * lda + k0);
                unsigned ldsa = (unsigned)(size_t)(&As[0][0]);
                u32x4 g0;
                g0[0] = 1u;                                        // count=1 user D#
                g0[1] = ldsa;                                      // lds_addr
                g0[2] = (unsigned)(ga & 0xffffffffull);            // global_addr lo
                g0[3] = (unsigned)((ga >> 32) & 0x1ffffffull)
                      | (2u << 30);                                // type=2 (image)
                i32x8 g1;
                g1[0] = 0x00010000;          // wg_mask=0, data_size=1 (2 bytes)
                g1[1] = 0;                   // barrier addr=0, tensor_dim0 lo16=0
                g1[2] = 0x4000;              // tensor_dim0=2^30, tensor_dim1 lo16=0
                g1[3] = 0x4000 | (BK << 16); // tensor_dim1=2^30, tile_dim0=BK
                g1[4] = BM;                  // tile_dim1=BM, tile_dim2=0
                g1[5] = lda;                 // tensor_dim0_stride lo32 (elements)
                g1[6] = 0;
                g1[7] = 0;
                i32x4 g2 = {}; i32x4 g3 = {};
#if __clang_major__ >= 23
                i32x8 g4 = {};
                __builtin_amdgcn_tensor_load_to_lds(g0, g1, g2, g3, g4, 0);
#else
                __builtin_amdgcn_tensor_load_to_lds(g0, g1, g2, g3, 0);
#endif
            }
#else
            #pragma unroll
            for (int i = 0; i < ACH; ++i) {
                int cid = tid + i * 256;
                int r = cid >> 2, c = (cid & 3) * 8;
                *(v8bf*)(&As[r][c]) =
                    *(const v8bf*)(Abfb + (long long)(m0 + r) * lda + k0 + c);
            }
#endif
        } else {
            #pragma unroll
            for (int i = 0; i < ANV; ++i) {
                int cid = tid + i * 256;
                int r = cid / KV4, c = (cid % KV4) * 4;
                v4f v = *(const v4f*)(Ab + (long long)(m0 + r) * lda + (k0 + c));
                As[r][c + 0] = f2bf(v[0]); As[r][c + 1] = f2bf(v[1]);
                As[r][c + 2] = f2bf(v[2]); As[r][c + 3] = f2bf(v[3]);
            }
        }
        // ---- B (weights) tile -> LDS, fp32 load + bf16 convert ----
        #pragma unroll
        for (int i = 0; i < BNV; ++i) {
            int cid = tid + i * 256;
            int r = cid / KV4, c = (cid % KV4) * 4;
            v4f v = *(const v4f*)(Wb + (long long)(n0 + r) * ldw + (k0 + c));
            Bs[r][c + 0] = f2bf(v[0]); Bs[r][c + 1] = f2bf(v[1]);
            Bs[r][c + 2] = f2bf(v[2]); Bs[r][c + 3] = f2bf(v[3]);
        }
        // prefetch next K tile of the weight stream (global_prefetch_b8)
        if (k0 + BK < K) {
            __builtin_prefetch(Wb + (long long)(n0 + (tid % BN)) * ldw + (k0 + BK), 0, 1);
            if constexpr (!ABF)
                __builtin_prefetch(Ab + (long long)(m0 + (tid % BM)) * lda + (k0 + BK), 0, 1);
        }
#ifdef HAVE_TDM
        if constexpr (ABF) {
            if (tid == 0) {
#if __has_builtin(__builtin_amdgcn_s_wait_tensorcnt)
                __builtin_amdgcn_s_wait_tensorcnt((short)0);
#endif
            }
        }
#endif
        __syncthreads();

        // ---- fragment loads (ds_load_b128) + WMMA ----
        v16bf af[MI];
        v16bf bfr[NI];
        #pragma unroll
        for (int mi = 0; mi < MI; ++mi) {
            int row = wm * WM + mi * 16 + lm;
            const __bf16* ap = &As[row][kbA];
            AFrag u;
            u.h[0] = *(const v8bf*)(ap);        // K = kb .. kb+7
            u.h[1] = *(const v8bf*)(ap + 8);    // K = kb+8 .. kb+15
            af[mi] = u.v;
        }
        #pragma unroll
        for (int ni = 0; ni < NI; ++ni) {
            int col = wn * WN + ni * 16 + lm;
            bfr[ni] = *(const v16bf*)(&Bs[col][kbB]);   // K = kbB .. kbB+15
        }
        #pragma unroll
        for (int mi = 0; mi < MI; ++mi)
            #pragma unroll
            for (int ni = 0; ni < NI; ++ni)
                acc[mi][ni] = __builtin_amdgcn_wmma_f32_16x16x32_bf16(
                    false, af[mi], false, bfr[ni], (short)0, acc[mi][ni],
                    false, false);
        __syncthreads();
    }

    // ---- epilogue ----
    #pragma unroll
    for (int mi = 0; mi < MI; ++mi) {
        #pragma unroll
        for (int ni = 0; ni < NI; ++ni) {
            const int tileM = m0 + wm * WM + mi * 16;
            const int n     = n0 + wn * WN + ni * 16 + lm;
            #pragma unroll
            for (int r = 0; r < 8; ++r) {
                const int m = tileM + r + 8 * lg;   // C/D layout: VGPR r -> M=r / r+8
                float v = acc[mi][ni][r];
                if constexpr (EPI == EPI_BIAS) {
                    v += bias[biasOff + n];
                    Out[obase + (long long)m * ldo + n] = v;
                    OutB[obase + (long long)m * ldo + n] = f2bf(v);
                } else if constexpr (EPI == EPI_BIAS_RELU) {
                    v = fmaxf(v + bias[biasOff + n], 0.0f);
                    OutB[obase + (long long)m * ldo + n] = f2bf(v);
                } else if constexpr (EPI == EPI_BIAS_RES) {
                    v += bias[biasOff + n] + Res[(long long)m * ldr + n];
                    Out[obase + (long long)m * ldo + n] = v;
                } else if constexpr (EPI == EPI_QKHEAD) {
                    v += bias[biasOff + n];
                    long long idx = (long long)(m / SEQ) * (NH * SEQ * DH)
                                  + (long long)(n / DH) * (SEQ * DH)
                                  + (long long)(m % SEQ) * DH + (n % DH);
                    Out[idx] = v;
                } else if constexpr (EPI == EPI_VTHEAD) {
                    v += bias[biasOff + n];
                    long long idx = (long long)(m / SEQ) * (NH * DH * SEQ)
                                  + (long long)(n / DH) * (DH * SEQ)
                                  + (long long)(n % DH) * SEQ + (m % SEQ);
                    Out[idx] = v;
                } else if constexpr (EPI == EPI_SCALE) {
                    Out[obase + (long long)m * ldo + n] = v * alpha;
                } else if constexpr (EPI == EPI_PLAIN) {
                    OutB[obase + (long long)m * ldo + n] = f2bf(v);
                } else { // EPI_MASK
                    Out[obase + (long long)m * ldo + n] =
                        (v != 0.0f) ? Res[(long long)m * ldr + n] : -1e10f;
                }
            }
        }
    }
}

// ---------------------------------------------------------------------------
// Embedding + positional add (dual fp32 + bf16 output)
// ---------------------------------------------------------------------------
__launch_bounds__(256)
__global__ void embed_kernel(const int* __restrict__ tok,
                             const float* __restrict__ emb,
                             const float* __restrict__ pe,
                             float* __restrict__ out,
                             __bf16* __restrict__ outb, int total)
{
    int i = blockIdx.x * 256 + threadIdx.x;
    if (i >= total) return;
    int d = i % DM;
    int p = i / DM;          // b*SEQ + s
    int s = p % SEQ;
    float v = emb[(long long)tok[p] * DM + d] + pe[s * DM + d];
    out[i] = v;
    outb[i] = f2bf(v);
}

// ---------------------------------------------------------------------------
// Row LayerNorm over D columns, one block per row (dual fp32 + bf16 output)
// ---------------------------------------------------------------------------
__launch_bounds__(256)
__global__ void layernorm_kernel(const float* __restrict__ in,
                                 float* __restrict__ out,
                                 __bf16* __restrict__ outb,
                                 const float* __restrict__ g,
                                 const float* __restrict__ b, int D)
{
    __shared__ float s1[256], s2[256];
    const int row = blockIdx.x, tid = threadIdx.x;
    const float* p = in + (long long)row * D;
    float sum = 0.0f, sq = 0.0f;
    for (int i = tid; i < D; i += 256) { float v = p[i]; sum += v; sq += v * v; }
    s1[tid] = sum; s2[tid] = sq;
    __syncthreads();
    for (int o = 128; o > 0; o >>= 1) {
        if (tid < o) { s1[tid] += s1[tid + o]; s2[tid] += s2[tid + o]; }
        __syncthreads();
    }
    const float mean = s1[0] / (float)D;
    const float var  = s2[0] / (float)D - mean * mean;
    const float inv  = rsqrtf(var + 1e-5f);
    float*  q  = out  + (long long)row * D;
    __bf16* qb = outb + (long long)row * D;
    for (int i = tid; i < D; i += 256) {
        float v = (p[i] - mean) * inv * g[i] + b[i];
        q[i] = v;
        qb[i] = f2bf(v);
    }
}

// ---------------------------------------------------------------------------
// In-place row softmax over W columns, one block per row (256 threads)
// ---------------------------------------------------------------------------
__launch_bounds__(256)
__global__ void softmax_kernel(float* __restrict__ buf, int W)
{
    __shared__ float sm[256];
    const int row = blockIdx.x, tid = threadIdx.x;
    float* p = buf + (long long)row * W;
    float mx = -3.4e38f;
    for (int i = tid; i < W; i += 256) mx = fmaxf(mx, p[i]);
    sm[tid] = mx;
    __syncthreads();
    for (int o = 128; o > 0; o >>= 1) {
        if (tid < o) sm[tid] = fmaxf(sm[tid], sm[tid + o]);
        __syncthreads();
    }
    mx = sm[0];
    __syncthreads();
    float sum = 0.0f;
    for (int i = tid; i < W; i += 256) { float e = __expf(p[i] - mx); p[i] = e; sum += e; }
    sm[tid] = sum;
    __syncthreads();
    for (int o = 128; o > 0; o >>= 1) {
        if (tid < o) sm[tid] += sm[tid + o];
        __syncthreads();
    }
    const float inv = 1.0f / sm[0];
    for (int i = tid; i < W; i += 256) p[i] *= inv;
}

// ---------------------------------------------------------------------------
// Host-side launch helper
// ---------------------------------------------------------------------------
template <int BM, int BN, int EPI, bool ABF>
static void launch_gemm(hipStream_t st,
                        const float* A, const __bf16* Abf, int lda, long long sA,
                        const float* W, int ldw, long long sW,
                        const float* bias, int biasOff,
                        float* Out, __bf16* OutB, int ldo,
                        int bInner, long long soO, long long soI,
                        const float* Res, int ldr,
                        int M, int N, int K, float alpha, int batch)
{
    dim3 grid(M / BM, N / BN, batch), blk(256);
    gemm_nt_wmma<BM, BN, EPI, ABF><<<grid, blk, 0, st>>>(
        A, Abf, lda, sA, W, ldw, sW, bias, biasOff, Out, OutB, ldo,
        bInner, soO, soI, Res, ldr, M, N, K, alpha);
}

// Full MHA: o = softmax(QK^T/sqrt(dh)) V  -> bf16 out laid out [B,S,D]
static void attention(hipStream_t st, const __bf16* xqb, const __bf16* xkvb,
                      const float* qkv_w, const float* qkv_b,
                      float* q, float* k, float* vt, float* sc, __bf16* hb)
{
    // Q = xq Wq^T + bq   -> [b,h,s,dh]  (A from bf16 activations via TDM)
    launch_gemm<128, 128, EPI_QKHEAD, true>(st, nullptr, xqb, DM, 0,
        qkv_w, DM, 0, qkv_b, 0, q, nullptr, 0, 1, 0, 0, nullptr, 0,
        MTOK, DM, DM, 1.0f, 1);
    // K = xkv Wk^T + bk  -> [b,h,s,dh]
    launch_gemm<128, 128, EPI_QKHEAD, true>(st, nullptr, xkvb, DM, 0,
        qkv_w + (long long)DM * DM, DM, 0, qkv_b, DM, k, nullptr, 0,
        1, 0, 0, nullptr, 0, MTOK, DM, DM, 1.0f, 1);
    // V^T = (xkv Wv^T + bv)^T -> [b,h,dh,s]
    launch_gemm<128, 128, EPI_VTHEAD, true>(st, nullptr, xkvb, DM, 0,
        qkv_w + 2LL * DM * DM, DM, 0, qkv_b, 2 * DM, vt, nullptr, 0,
        1, 0, 0, nullptr, 0, MTOK, DM, DM, 1.0f, 1);
    // scores[z=b*NH+h] = Q K^T * 1/8  (fp32-A path)
    launch_gemm<128, 128, EPI_SCALE, false>(st, q, nullptr, DH, (long long)SEQ * DH,
        k, DH, (long long)SEQ * DH, nullptr, 0, sc, nullptr, SEQ,
        1, (long long)SEQ * SEQ, 0, nullptr, 0, SEQ, SEQ, DH, 0.125f, BB * NH);
    softmax_kernel<<<BB * NH * SEQ, 256, 0, st>>>(sc, SEQ);
    // o[z] = P (V^T)^T -> bf16, scattered into [B,S,D]
    launch_gemm<64, 64, EPI_PLAIN, false>(st, sc, nullptr, SEQ, (long long)SEQ * SEQ,
        vt, SEQ, (long long)DH * SEQ, nullptr, 0, nullptr, hb, DM,
        NH, (long long)SEQ * DM, DH, nullptr, 0, SEQ, DH, SEQ, 1.0f, BB * NH);
}

// ---------------------------------------------------------------------------
// Entry point
// ---------------------------------------------------------------------------
extern "C" void kernel_launch(void* const* d_in, const int* in_sizes, int n_in,
                              void* d_out, int out_size, void* d_ws, size_t ws_size,
                              hipStream_t stream)
{
    (void)in_sizes; (void)n_in; (void)out_size; (void)ws_size;

    const int*   src  = (const int*)d_in[0];
    const int*   tgt  = (const int*)d_in[1];
    const float* emb  = (const float*)d_in[2];
    const float* pos  = (const float*)d_in[3];
    const float* mask = (const float*)d_in[4];
    const float* e_qkv_w  = (const float*)d_in[5];
    const float* e_qkv_b  = (const float*)d_in[6];
    const float* e_proj_w = (const float*)d_in[7];
    const float* e_proj_b = (const float*)d_in[8];
    const float* e_ff1_w  = (const float*)d_in[9];
    const float* e_ff1_b  = (const float*)d_in[10];
    const float* e_ff2_w  = (const float*)d_in[11];
    const float* e_ff2_b  = (const float*)d_in[12];
    const float* e_ln1_g  = (const float*)d_in[13];
    const float* e_ln1_b  = (const float*)d_in[14];
    const float* e_ln2_g  = (const float*)d_in[15];
    const float* e_ln2_b  = (const float*)d_in[16];
    const float* e_nrm_g  = (const float*)d_in[17];
    const float* e_nrm_b  = (const float*)d_in[18];
    const float* s_qkv_w  = (const float*)d_in[19];
    const float* s_qkv_b  = (const float*)d_in[20];
    const float* s_proj_w = (const float*)d_in[21];
    const float* s_proj_b = (const float*)d_in[22];
    const float* c_qkv_w  = (const float*)d_in[23];
    const float* c_qkv_b  = (const float*)d_in[24];
    const float* c_proj_w = (const float*)d_in[25];
    const float* c_proj_b = (const float*)d_in[26];
    const float* f_ff1_w  = (const float*)d_in[27];
    const float* f_ff1_b  = (const float*)d_in[28];
    const float* f_ff2_w  = (const float*)d_in[29];
    const float* f_ff2_b  = (const float*)d_in[30];
    const float* d_ln1_g  = (const float*)d_in[31];
    const float* d_ln1_b  = (const float*)d_in[32];
    const float* d_ln2_g  = (const float*)d_in[33];
    const float* d_ln2_b  = (const float*)d_in[34];
    const float* d_ln3_g  = (const float*)d_in[35];
    const float* d_ln3_b  = (const float*)d_in[36];
    const float* d_nrm_g  = (const float*)d_in[37];
    const float* d_nrm_b  = (const float*)d_in[38];
    const float* fc_w     = (const float*)d_in[39];
    const float* fc_b     = (const float*)d_in[40];

    // Workspace layout: fp32 region (~76 MB) then bf16 region (~22 MB)
    float* ws = (float*)d_ws;
    const long long MT = (long long)MTOK * DM;               // 1M elements
    float* x    = ws;                                        // [B,S,D]
    float* y    = x + MT;
    float* memb = y + MT;
    float* q    = memb + MT;                                 // [B,H,S,DH]
    float* k    = q + MT;
    float* vt   = k + MT;                                    // [B,H,DH,S]
    float* sc   = vt + MT;                                   // [B,H,S,S]   8M
    float* t1   = sc + (long long)BB * NH * SEQ * SEQ;       // [B,S,D]     1M
    float* lgts = t1 + MT;                                   // [B,S,V]     4M
    __bf16* xb    = (__bf16*)(lgts + (long long)MTOK * VOC); // [B,S,D]
    __bf16* yb    = xb + MT;
    __bf16* membb = yb + MT;
    __bf16* hb    = membb + MT;                              // [B,S,DFF]
    __bf16* lgb   = hb + (long long)MTOK * DFF;              // [B,S,V]
    float* outp = (float*)d_out;

    const int total = MTOK * DM;
    embed_kernel<<<(total + 255) / 256, 256, 0, stream>>>(src, emb, pos, x, xb, total);
    embed_kernel<<<(total + 255) / 256, 256, 0, stream>>>(tgt, emb, pos, y, yb, total);

    // ---------------- encoder ----------------
    for (int i = 0; i < NL; ++i) {
        attention(stream, xb, xb,
                  e_qkv_w + (long long)i * 3 * DM * DM,
                  e_qkv_b + (long long)i * 3 * DM, q, k, vt, sc, hb);
        launch_gemm<128, 128, EPI_BIAS_RES, true>(stream, nullptr, hb, DM, 0,
            e_proj_w + (long long)i * DM * DM, DM, 0, e_proj_b + (long long)i * DM, 0,
            t1, nullptr, DM, 1, 0, 0, x, DM, MTOK, DM, DM, 1.0f, 1);
        layernorm_kernel<<<MTOK, 256, 0, stream>>>(t1, x, xb,
            e_ln1_g + (long long)i * DM, e_ln1_b + (long long)i * DM, DM);
        launch_gemm<128, 128, EPI_BIAS_RELU, true>(stream, nullptr, xb, DM, 0,
            e_ff1_w + (long long)i * DFF * DM, DM, 0, e_ff1_b + (long long)i * DFF, 0,
            nullptr, hb, DFF, 1, 0, 0, nullptr, 0, MTOK, DFF, DM, 1.0f, 1);
        launch_gemm<128, 128, EPI_BIAS_RES, true>(stream, nullptr, hb, DFF, 0,
            e_ff2_w + (long long)i * DM * DFF, DFF, 0, e_ff2_b + (long long)i * DM, 0,
            t1, nullptr, DM, 1, 0, 0, x, DM, MTOK, DM, DFF, 1.0f, 1);
        layernorm_kernel<<<MTOK, 256, 0, stream>>>(t1, x, xb,
            e_ln2_g + (long long)i * DM, e_ln2_b + (long long)i * DM, DM);
    }
    layernorm_kernel<<<MTOK, 256, 0, stream>>>(x, memb, membb, e_nrm_g, e_nrm_b, DM);

    // ---------------- decoder ----------------
    for (int i = 0; i < NL; ++i) {
        // self-attention
        attention(stream, yb, yb,
                  s_qkv_w + (long long)i * 3 * DM * DM,
                  s_qkv_b + (long long)i * 3 * DM, q, k, vt, sc, hb);
        launch_gemm<128, 128, EPI_BIAS_RES, true>(stream, nullptr, hb, DM, 0,
            s_proj_w + (long long)i * DM * DM, DM, 0, s_proj_b + (long long)i * DM, 0,
            t1, nullptr, DM, 1, 0, 0, y, DM, MTOK, DM, DM, 1.0f, 1);
        layernorm_kernel<<<MTOK, 256, 0, stream>>>(t1, y, yb,
            d_ln1_g + (long long)i * DM, d_ln1_b + (long long)i * DM, DM);
        // cross-attention (K/V from encoder memory)
        attention(stream, yb, membb,
                  c_qkv_w + (long long)i * 3 * DM * DM,
                  c_qkv_b + (long long)i * 3 * DM, q, k, vt, sc, hb);
        launch_gemm<128, 128, EPI_BIAS_RES, true>(stream, nullptr, hb, DM, 0,
            c_proj_w + (long long)i * DM * DM, DM, 0, c_proj_b + (long long)i * DM, 0,
            t1, nullptr, DM, 1, 0, 0, y, DM, MTOK, DM, DM, 1.0f, 1);
        layernorm_kernel<<<MTOK, 256, 0, stream>>>(t1, y, yb,
            d_ln2_g + (long long)i * DM, d_ln2_b + (long long)i * DM, DM);
        // ffn
        launch_gemm<128, 128, EPI_BIAS_RELU, true>(stream, nullptr, yb, DM, 0,
            f_ff1_w + (long long)i * DFF * DM, DM, 0, f_ff1_b + (long long)i * DFF, 0,
            nullptr, hb, DFF, 1, 0, 0, nullptr, 0, MTOK, DFF, DM, 1.0f, 1);
        launch_gemm<128, 128, EPI_BIAS_RES, true>(stream, nullptr, hb, DFF, 0,
            f_ff2_w + (long long)i * DM * DFF, DFF, 0, f_ff2_b + (long long)i * DM, 0,
            t1, nullptr, DM, 1, 0, 0, y, DM, MTOK, DM, DFF, 1.0f, 1);
        layernorm_kernel<<<MTOK, 256, 0, stream>>>(t1, y, yb,
            d_ln3_g + (long long)i * DM, d_ln3_b + (long long)i * DM, DM);
    }
    layernorm_kernel<<<MTOK, 256, 0, stream>>>(y, x, xb, d_nrm_g, d_nrm_b, DM);

    // ---------------- vocab projection + logit mask ----------------
    // logits = out @ fc_w^T + fc_b   (dual fp32 + bf16 write)
    launch_gemm<128, 128, EPI_BIAS, true>(stream, nullptr, xb, DM, 0,
        fc_w, DM, 0, fc_b, 0, lgts, lgb, VOC, 1, 0, 0, nullptr, 0,
        MTOK, VOC, DM, 1.0f, 1);
    // result[m,i] = sum_j mask[i,j] logits[m,j];  out = result!=0 ? logits : -1e10
    launch_gemm<128, 128, EPI_MASK, true>(stream, nullptr, lgb, VOC, 0,
        mask, VOC, 0, nullptr, 0, outp, nullptr, VOC, 1, 0, 0, lgts, VOC,
        MTOK, VOC, VOC, 1.0f, 1);
}
